// InstanceConv_55095840473188
// MI455X (gfx1250) — matrix operand order repaired and checked by reading, not compiled, not run on hardware
//
#include <hip/hip_runtime.h>

// ---------------------------------------------------------------------------
// Masked ("instance") 3x3 conv as implicit GEMM on CDNA5 WMMA (bf16 -> f32).
//   B=8, C_in=128, C_out=256, H=W=128, K=3, pad=1, stride=1
//   out = ( sum_taps  W_tap @ (x_tap * eq_tap) ) * 9/(count+eps) + bias
// Mask gating is applied by *address selection*: masked-off taps read a
// zeroed LDS pad, so the WMMA contribution is exactly zero (1 cndmask per
// fragment instead of 8).
// ---------------------------------------------------------------------------

typedef __attribute__((ext_vector_type(16))) __bf16 v16bf;
typedef __attribute__((ext_vector_type(8)))  float  v8f;

#define CIN   128
#define COUT  256
#define HH    128
#define WW    128
#define TAPS  9
#define KCH   4        // K chunks of 32 channels (128/32)
#define NPIX  64       // pixels per workgroup (strip width)
#define XELEMS (66 * 3 * CIN)            // 25344 halo elements
#define ZPAD   128                        // zeroed pad (covers kc*32 + kb16)
#define OUT_ELEMS (8 * COUT * HH * WW)   // 33,554,432 f32
#define MASK_ELEMS (8 * HH * WW)         // 131,072 i32

// ---------------------------------------------------------------------------
// Pack fp32 weights (C_out, C_in, 3, 3) into bf16 fragments matching the
// 16x32 A-fragment order used here (contiguous K per lane half, as in B), so each lane's whole
// fragment is one contiguous 32-byte read.
//   Layout: pw[ ((t*4+kc)*16 + msub)*32 + lane ][ j(0..15) ]
//   lane<16: K = {0..15} ; lane>=16: K = {16..31}
// ---------------------------------------------------------------------------
__global__ __launch_bounds__(256) void pack_weights_kernel(
    const float* __restrict__ w, unsigned short* __restrict__ pw_raw) {
  __bf16* pw = (__bf16*)pw_raw;
  int idx = blockIdx.x * 256 + threadIdx.x;          // 0 .. 294911
  int j    =  idx        & 15;
  int lane = (idx >> 4)  & 31;
  int msub = (idx >> 9)  & 15;
  int kc   = (idx >> 13) & 3;
  int t    =  idx >> 15;
  int kb = (lane < 16) ? 0 : 16;
  int kl = kb + j;
  int cin  = kc * 32 + kl;
  int cout = msub * 16 + (lane & 15);
  int di = t / 3, dj = t % 3;
  float v = w[((cout * CIN + cin) * 3 + di) * 3 + dj];
  pw[idx] = (__bf16)v;
}

// ---------------------------------------------------------------------------
// Main kernel: one block = (b, h, 64-pixel strip). 8 wave32 per block.
// Wave w owns M-subtiles {2w, 2w+1} (32 out-channels) x 4 N-subtiles.
// 288 v_wmma_f32_16x16x32_bf16 per wave.
// ---------------------------------------------------------------------------
__global__ __launch_bounds__(256) void instconv_wmma_kernel(
    const float* __restrict__ inp, const int* __restrict__ mask,
    const unsigned short* __restrict__ pw_raw, const float* __restrict__ bias,
    float* __restrict__ out) {
  const __bf16* pw = (const __bf16*)pw_raw;

  // LDS: halo input in bf16, [col 0..65][row 0..2][c 0..127] (channel fastest)
  // plus a zeroed 128-element pad used as the "masked-off" fragment source.
  __shared__ __align__(32) __bf16 lds_x[XELEMS + ZPAD];  // 50944 B
  __shared__ float    s_count[NPIX];
  __shared__ unsigned s_eqm[NPIX];
  __shared__ float    s_bias[COUT];

  const int tid  = threadIdx.x;
  const int lane = tid & 31;
  const int wave = tid >> 5;

  const int blk = blockIdx.x;               // 2048 blocks
  const int w0  = (blk & 1) * NPIX;         // strip start col
  const int h   = (blk >> 1) & (HH - 1);
  const int b   =  blk >> 8;

  // ---- stage bias + zero pad ----
  s_bias[tid] = bias[tid];                  // tid in [0,256)
  if (tid < ZPAD) lds_x[XELEMS + tid] = (__bf16)0.0f;

  // ---- stage eq bits + count (one thread per pixel) ----
  if (tid < NPIX) {
    const int p = tid;
    const int center = mask[(b * HH + h) * WW + (w0 + p)];
    unsigned bits = 0; int cnt = 0;
    #pragma unroll
    for (int t = 0; t < TAPS; ++t) {
      const int di = t / 3, dj = t % 3;
      const int hh = h + di - 1;
      const int ww = w0 + p + dj - 1;
      int m = -1;
      if (hh >= 0 && hh < HH && ww >= 0 && ww < WW)
        m = mask[(b * HH + hh) * WW + ww];
      if (m == center) { bits |= (1u << t); ++cnt; }
    }
    s_eqm[p]   = bits;
    s_count[p] = (float)cnt;
  }

  // ---- stage input halo, fp32 -> bf16 (padded positions are masked out) ----
  for (int i = tid; i < XELEMS; i += 256) {
    const int c   = i & (CIN - 1);
    const int rc  = i >> 7;
    const int row = rc % 3;
    const int col = rc / 3;
    const int hh  = h  + row - 1;
    const int ww  = w0 + col - 1;
    float v = 0.0f;
    if (hh >= 0 && hh < HH && ww >= 0 && ww < WW)
      v = inp[((b * CIN + c) * HH + hh) * WW + ww];
    lds_x[i] = (__bf16)v;
  }
  __syncthreads();

  // ---- per-lane loop-invariants ----
  const int pixbase = lane & 15;            // N within subtile
  const int kb16    = (lane < 16) ? 0 : 16; // B-fragment K-half
  unsigned em[4]; float cnt[4];
  #pragma unroll
  for (int n = 0; n < 4; ++n) {
    em[n]  = s_eqm[n * 16 + pixbase];
    cnt[n] = s_count[n * 16 + pixbase];
  }

  v8f acc[2][4];
  #pragma unroll
  for (int mi = 0; mi < 2; ++mi)
    #pragma unroll
    for (int n = 0; n < 4; ++n)
      #pragma unroll
      for (int r = 0; r < 8; ++r) acc[mi][n][r] = 0.0f;

  const v16bf* pwf = (const v16bf*)pw;
  const int msub0 = wave * 2;

  // ---- main GEMM loop: 9 taps x 4 K-chunks ----
  for (int t = 0; t < TAPS; ++t) {
    const int di = t / 3, dj = t % 3;

    // Address-select gating: one select per (tap, N-subtile). Masked-off
    // fragments read the zeroed LDS pad (kc*32 + kb16 stays within ZPAD).
    int nbase[4];
    #pragma unroll
    for (int n = 0; n < 4; ++n) {
      const int pix  = n * 16 + pixbase;
      const int base = ((pix + dj) * 3 + di) * CIN;
      const bool on  = (em[n] >> t) & 1u;
      nbase[n] = on ? base : XELEMS;
    }

    #pragma unroll
    for (int kc = 0; kc < KCH; ++kc) {
      const int fbase = ((t * KCH + kc) * 16 + msub0) * 32 + lane;
      const v16bf a0 = pwf[fbase];
      const v16bf a1 = pwf[fbase + 32];
      // prefetch next A tile (L2-resident weights -> WGP$)
      __builtin_prefetch((const void*)(pwf + fbase + 16 * 32), 0, 0);

      #pragma unroll
      for (int n = 0; n < 4; ++n) {
        const v16bf bf = *(const v16bf*)&lds_x[nbase[n] + kc * 32 + kb16];
        acc[0][n] = __builtin_amdgcn_wmma_f32_16x16x32_bf16(
            false, a0, false, bf, (short)0, acc[0][n], false, false);
        acc[1][n] = __builtin_amdgcn_wmma_f32_16x16x32_bf16(
            false, a1, false, bf, (short)0, acc[1][n], false, false);
      }
    }
  }

  // ---- epilogue: scale by 9/(count+eps), add bias, store fp32 ----
  const int madd = (lane >= 16) ? 8 : 0;    // C/D layout: VGPR r -> M = r(+8)
  #pragma unroll
  for (int n = 0; n < 4; ++n) {
    const float norm = 9.0f / (cnt[n] + 1e-5f);
    const int wcol = w0 + n * 16 + pixbase;
    #pragma unroll
    for (int r = 0; r < 8; ++r) {
      const int m  = r + madd;
      const int c0 = (msub0 + 0) * 16 + m;
      const int c1 = (msub0 + 1) * 16 + m;
      out[((b * COUT + c0) * HH + h) * WW + wcol] = acc[0][n][r] * norm + s_bias[c0];
      out[((b * COUT + c1) * HH + h) * WW + wcol] = acc[1][n][r] * norm + s_bias[c1];
    }
  }
}

// ---------------------------------------------------------------------------
// Second tuple output: mask_out = center mask = original mask (int32),
// concatenated after the fp32 conv output.
// ---------------------------------------------------------------------------
__global__ __launch_bounds__(256) void copy_mask_kernel(
    const int* __restrict__ mask, int* __restrict__ mout) {
  int i = blockIdx.x * 256 + threadIdx.x;
  if (i < MASK_ELEMS) mout[i] = mask[i];
}

extern "C" void kernel_launch(void* const* d_in, const int* in_sizes, int n_in,
                              void* d_out, int out_size, void* d_ws, size_t ws_size,
                              hipStream_t stream) {
  const float* inp    = (const float*)d_in[0];
  const int*   mask   = (const int*)d_in[1];
  const float* weight = (const float*)d_in[2];
  const float* bias   = (const float*)d_in[3];

  unsigned short* pw = (unsigned short*)d_ws;              // 576 KB bf16 weights
  float* out  = (float*)d_out;
  int*   mout = (int*)d_out + OUT_ELEMS;

  // 1) pack weights fp32 -> bf16 WMMA A-fragment layout (294912 elems)
  pack_weights_kernel<<<294912 / 256, 256, 0, stream>>>(weight, pw);
  // 2) implicit-GEMM masked conv: 8 batches * 128 rows * 2 strips = 2048 blocks
  instconv_wmma_kernel<<<2048, 256, 0, stream>>>(inp, mask, pw, bias, out);
  // 3) mask_out passthrough
  copy_mask_kernel<<<(MASK_ELEMS + 255) / 256, 256, 0, stream>>>(mask, mout);
}
